// Conv1d_NN_Attn_spatial_20435454394586
// MI455X (gfx1250) — matrix operand order repaired and checked by reading, not compiled, not run on hardware
//
#include <hip/hip_runtime.h>
#include <stdint.h>

// ---------------------------------------------------------------------------
// Problem constants (match the JAX reference)
// ---------------------------------------------------------------------------
#define BB  32
#define CC  64
#define TT  4096
#define SS  512
#define KK  9
#define OCC 64

typedef __attribute__((ext_vector_type(16))) __bf16        v16bf;
typedef __attribute__((ext_vector_type(8)))  float         v8f;
typedef __attribute__((ext_vector_type(8)))  unsigned int  v8u;
typedef __attribute__((ext_vector_type(4)))  unsigned int  v4u;
typedef __attribute__((ext_vector_type(8)))  int           v8i_t;
typedef __attribute__((ext_vector_type(4)))  int           v4i_t;

// float -> bf16 with round-to-nearest-even (pure integer math, no __bf16 ops)
__device__ __forceinline__ unsigned short f2bf(float f) {
  union { float f; unsigned int u; } cv; cv.f = f;
  unsigned int u = cv.u;
  return (unsigned short)((u + 0x7FFFu + ((u >> 16) & 1u)) >> 16);
}

// ---------------------------------------------------------------------------
// WMMA fragment loaders for the NT (A[M,K] row-major, B[N,K] row-major) case.
// A 16x32 bf16 layout: lanes 0-15 hold M=lane, K 0..7 (v0-3) and 16..23 (v4-7)
//                      lanes 16-31 hold M=lane-16, K 8..15 and 24..31
// B 32x16 bf16 layout: lane%16 = N column, lanes 0-15 hold K 0..7 and 16..23,
//                      lanes 16-31 hold K 8..15 and 24..31 (same as A)
// Work for both global and LDS-backed pointers (16B-aligned uint4 accesses).
// ---------------------------------------------------------------------------
__device__ __forceinline__ v16bf load_frag_a(const unsigned short* base,
                                             int ld, int row0, int col0, int lane) {
  int r = row0 + (lane & 15);
  const unsigned short* p = base + (size_t)r * ld + col0 + ((lane >> 4) << 3);
  uint4 lo = *(const uint4*)(p);        // K +0..7   (or +8..15)
  uint4 hi = *(const uint4*)(p + 16);   // K +16..23 (or +24..31)
  v8u v;
  v[0] = lo.x; v[1] = lo.y; v[2] = lo.z; v[3] = lo.w;
  v[4] = hi.x; v[5] = hi.y; v[6] = hi.z; v[7] = hi.w;
  return __builtin_bit_cast(v16bf, v);
}

__device__ __forceinline__ v16bf load_frag_b(const unsigned short* base,
                                             int ld, int n0, int k0, int lane) {
  int r = n0 + (lane & 15);
  const unsigned short* p = base + (size_t)r * ld + k0 + ((lane >> 4) << 3);
  uint4 lo = *(const uint4*)(p);        // K +0..7   (or +8..15)
  uint4 hi = *(const uint4*)(p + 16);   // K +16..23 (or +24..31)
  v8u v;
  v[0] = lo.x; v[1] = lo.y; v[2] = lo.z; v[3] = lo.w;
  v[4] = hi.x; v[5] = hi.y; v[6] = hi.z; v[7] = hi.w;
  return __builtin_bit_cast(v16bf, v);
}

__device__ __forceinline__ v8f wmma_bf16(v16bf a, v16bf b, v8f c) {
  // D = A*B + C, f32 accumulate  ->  v_wmma_f32_16x16x32_bf16
  return __builtin_amdgcn_wmma_f32_16x16x32_bf16(false, a, false, b,
                                                 (short)0, c, false, false);
}

// ---------------------------------------------------------------------------
// Tensor Data Mover: DMA a 2D tile [tile_h rows x tile_w elems] of bf16 from
// global memory into LDS (row-major compacted). D# bitfields per
// CDNA5 ISA. Issued once per wave (EXEC ignored);
// caller must gate to a single wave and track completion via TENSORcnt.
// This toolchain's builtin takes 6 args:
//   (uint32x4 g0, int32x8 g1, int32x4 g2, int32x4 g3, int32x8 pad, i32 cpol)
// ---------------------------------------------------------------------------
__device__ __forceinline__ void tdm_load_tile(const unsigned short* gsrc,
                                              unsigned lds_byte_off,
                                              unsigned tile_w, unsigned tile_h,
                                              unsigned tensor_w, unsigned tensor_h,
                                              unsigned stride_elems) {
  unsigned long long ga = (unsigned long long)(const void*)gsrc;
  v4u g0;
  g0[0] = 1u;                                             // count=1, user D#
  g0[1] = lds_byte_off;                                   // LDS byte address
  g0[2] = (unsigned)ga;                                   // global_addr[31:0]
  g0[3] = (unsigned)((ga >> 32) & 0x01FFFFFFu) | (2u << 30); // addr[56:32], type=2
  v8i_t g1;
  g1[0] = (int)(1u << 16);                                // data_size=1 (2 bytes)
  g1[1] = (int)((tensor_w & 0xFFFFu) << 16);              // tensor_dim0[15:0]
  g1[2] = (int)((tensor_w >> 16) | ((tensor_h & 0xFFFFu) << 16)); // dim0 hi | dim1 lo
  g1[3] = (int)((tensor_h >> 16) | (tile_w << 16));       // dim1 hi | tile_dim0
  g1[4] = (int)tile_h;                                    // tile_dim1 (tile_dim2=0)
  g1[5] = (int)stride_elems;                              // tensor_dim0_stride[31:0]
  g1[6] = 0;                                              // stride hi / dim1_stride lo
  g1[7] = 0;
  v4i_t gz4 = (v4i_t)0;                                   // groups 2/3 unused (2D)
  v8i_t gz8 = (v8i_t)0;
  __builtin_amdgcn_tensor_load_to_lds(g0, g1, gz4, gz4, gz8, 0);
}

// ---------------------------------------------------------------------------
// 1) elementwise f32 -> bf16 cast
// ---------------------------------------------------------------------------
__global__ void cast_f32_bf16(const float* __restrict__ in,
                              unsigned short* __restrict__ out, int n) {
  int i = blockIdx.x * blockDim.x + threadIdx.x;
  if (i < n) out[i] = f2bf(in[i]);
}

// ---------------------------------------------------------------------------
// 2) NT GEMM: C[M,N] = A[M,K] * B[N,K]^T, bf16 in, f32 out.
//    TDM-staged LDS tiles: block = 128 threads = 4 waves, block tile 64x64.
//    Wave 0 DMAs A[64x32] and B[64x32] tiles, waits TENSORcnt, barrier, then
//    each wave runs 4 WMMAs (A fragment reused across 4 N-subtiles) from LDS.
// ---------------------------------------------------------------------------
__global__ void gemm_nt(const unsigned short* __restrict__ A,
                        const unsigned short* __restrict__ Bw,
                        float* __restrict__ Cm,
                        int M, int N, int Kd) {
  __shared__ unsigned short ldsA[64 * 32];   // [row][k], 64B rows
  __shared__ unsigned short ldsB[64 * 32];
  int lane = threadIdx.x & 31;
  int wave = threadIdx.x >> 5;
  int m0 = blockIdx.y * 64;
  int n0 = blockIdx.x * 64;

  unsigned ldsA_off = (unsigned)(unsigned long long)(void*)ldsA;  // LDS aperture: low 32 bits
  unsigned ldsB_off = (unsigned)(unsigned long long)(void*)ldsB;

  v8f acc[4] = {{}, {}, {}, {}};
  for (int kk = 0; kk < Kd; kk += 32) {
    if (wave == 0) {   // wave-uniform branch: EXEC stays all-ones elsewhere
      tdm_load_tile(A  + (size_t)m0 * Kd + kk, ldsA_off, 32, 64,
                    (unsigned)Kd, (unsigned)M, (unsigned)Kd);
      tdm_load_tile(Bw + (size_t)n0 * Kd + kk, ldsB_off, 32, 64,
                    (unsigned)Kd, (unsigned)N, (unsigned)Kd);
      __builtin_amdgcn_s_wait_tensorcnt(0);  // s_wait_tensorcnt 0
    }
    __syncthreads();

    v16bf a = load_frag_a(ldsA, 32, wave * 16, 0, lane);
#pragma unroll
    for (int s = 0; s < 4; ++s) {
      v16bf b = load_frag_b(ldsB, 32, s * 16, 0, lane);
      acc[s] = wmma_bf16(a, b, acc[s]);
    }
    __syncthreads();   // tiles fully consumed before next DMA overwrites them
  }

  // D layout: VGPR r, lanes 0-15 -> M=r ; lanes 16-31 -> M=r+8 ; N = lane&15
  int row_off = (lane >> 4) << 3;
  int col     = lane & 15;
#pragma unroll
  for (int s = 0; s < 4; ++s) {
#pragma unroll
    for (int r = 0; r < 8; ++r) {
      int row = m0 + wave * 16 + r + row_off;
      Cm[(size_t)row * N + n0 + s * 16 + col] = acc[s][r];
    }
  }
}

// ---------------------------------------------------------------------------
// 3) channel-normalize over C and transpose [B,C,L] f32 -> [B,L,C] bf16
// ---------------------------------------------------------------------------
__global__ void norm_tr(const float* __restrict__ in,
                        unsigned short* __restrict__ out, int L) {
  int i = blockIdx.x * blockDim.x + threadIdx.x;
  if (i >= BB * L) return;
  int b = i / L, l = i - b * L;
  const float* src = in + (size_t)b * CC * L + l;
  float ss = 0.f;
#pragma unroll
  for (int c = 0; c < CC; ++c) { float t = src[(size_t)c * L]; ss += t * t; }
  float scale = 1.0f / fmaxf(sqrtf(ss), 1e-12f);
  unsigned short* dst = out + ((size_t)b * L + l) * CC;
#pragma unroll
  for (int c = 0; c < CC; ++c) dst[c] = f2bf(src[(size_t)c * L] * scale);
}

// ---------------------------------------------------------------------------
// 4) fused sim = relu(kt * qt^T) + running top-8 over S, per 16-token tile.
//    One wave per block; A fragments (the token tile) stay resident.
// ---------------------------------------------------------------------------
__global__ void sim_topk(const unsigned short* __restrict__ kt,   // [B,T,C] bf16
                         const unsigned short* __restrict__ qt,   // [B,S,C] bf16
                         const int* __restrict__ samp_idx,        // [S]
                         int* __restrict__ outIdx) {              // [B,T,KK]
  __shared__ float tile[16][16];
  int lane = threadIdx.x & 31;
  int b  = blockIdx.y;
  int t0 = blockIdx.x * 16;
  const unsigned short* ktb = kt + (size_t)b * TT * CC;
  const unsigned short* qtb = qt + (size_t)b * SS * CC;

  v16bf a0 = load_frag_a(ktb, CC, t0, 0,  lane);   // C = 64 -> two K-steps
  v16bf a1 = load_frag_a(ktb, CC, t0, 32, lane);

  float bestv[8]; int bests[8];
#pragma unroll
  for (int i = 0; i < 8; ++i) { bestv[i] = -1.0f; bests[i] = 0; }

  for (int s0 = 0; s0 < SS; s0 += 16) {
    v16bf b0 = load_frag_b(qtb, CC, s0, 0,  lane);
    v16bf b1 = load_frag_b(qtb, CC, s0, 32, lane);
    v8f acc = {};
    acc = wmma_bf16(a0, b0, acc);
    acc = wmma_bf16(a1, b1, acc);

    int half = lane >> 4, col = lane & 15;
#pragma unroll
    for (int r = 0; r < 8; ++r) tile[r + half * 8][col] = acc[r];
    __syncthreads();

    if (lane < 16) {              // lane owns token row t0+lane
#pragma unroll
      for (int j = 0; j < 16; ++j) {
        float val = fmaxf(tile[lane][j], 0.0f);      // relu
        if (val > bestv[7]) {                        // sorted-insert top-8
          int pos = 7;
          while (pos > 0 && val > bestv[pos - 1]) {
            bestv[pos] = bestv[pos - 1];
            bests[pos] = bests[pos - 1];
            --pos;
          }
          bestv[pos] = val;
          bests[pos] = s0 + j;
        }
      }
    }
    __syncthreads();
  }

  if (lane < 16) {
    int t = t0 + lane;
    int* o = outIdx + ((size_t)b * TT + t) * KK;
    o[0] = t;                                        // self token first
#pragma unroll
    for (int i = 0; i < 8; ++i) o[1 + i] = samp_idx[bests[i]];
  }
}

// ---------------------------------------------------------------------------
// 5) gather neighbors of v + conv1d(K=9, stride=9) + bias
//    block = 64 threads = one (b,t); thread = one output channel.
// ---------------------------------------------------------------------------
__global__ void gather_conv(const float* __restrict__ v,      // [B,C,T]
                            const int* __restrict__ nnIdx,    // [B,T,KK]
                            const float* __restrict__ w,      // [OC,C,KK]
                            const float* __restrict__ bias,   // [OC]
                            float* __restrict__ out) {        // [B,OC,T]
  __shared__ float prime[CC * KK];   // 576 gathered values for this (b,t)
  __shared__ int   toks[KK];
  int b = blockIdx.y;
  int t = blockIdx.x;
  int tid = threadIdx.x;
  const float* wr = w + (size_t)tid * CC * KK;
  __builtin_prefetch(wr, 0, 3);      // global_prefetch_b8 on the weight row
  if (tid < KK) toks[tid] = nnIdx[((size_t)b * TT + t) * KK + tid];
  __syncthreads();

  const float* vb = v + (size_t)b * CC * TT;
  for (int e = tid; e < CC * KK; e += OCC) {
    int c = e / KK, j = e - c * KK;
    prime[e] = vb[(size_t)c * TT + toks[j]];
  }
  __syncthreads();

  float acc = bias[tid];
#pragma unroll 9
  for (int e = 0; e < CC * KK; ++e) acc += wr[e] * prime[e];
  out[((size_t)b * OCC + tid) * TT + t] = acc;
}

// ---------------------------------------------------------------------------
// Host launcher. Scratch layout carved from d_ws (~181 MB total).
// ---------------------------------------------------------------------------
extern "C" void kernel_launch(void* const* d_in, const int* in_sizes, int n_in,
                              void* d_out, int out_size, void* d_ws, size_t ws_size,
                              hipStream_t stream) {
  const float* x    = (const float*)d_in[0];   // [B,C,T]
  const float* y    = (const float*)d_in[1];   // [B,C,S]
  const float* Wq   = (const float*)d_in[2];   // [S,S]
  const float* Wk   = (const float*)d_in[3];   // [T,T]
  const float* Wv   = (const float*)d_in[4];   // [T,T]
  const float* cw   = (const float*)d_in[5];   // [OC,C,KK]
  const float* cb   = (const float*)d_in[6];   // [OC]
  const int*   sidx = (const int*)d_in[7];     // [S]
  float* out = (float*)d_out;                  // [B,OC,T]

  char* ws = (char*)d_ws;
  size_t off = 0;
  auto take = [&](size_t bytes) -> char* {
    char* p = ws + off;
    off = (off + bytes + 255) & ~(size_t)255;
    return p;
  };

  unsigned short* xbf  = (unsigned short*)take((size_t)BB * CC * TT * 2);
  unsigned short* ybf  = (unsigned short*)take((size_t)BB * CC * SS * 2);
  unsigned short* wkbf = (unsigned short*)take((size_t)TT * TT * 2);
  unsigned short* wvbf = (unsigned short*)take((size_t)TT * TT * 2);
  unsigned short* wqbf = (unsigned short*)take((size_t)SS * SS * 2);
  float*          kf   = (float*)take((size_t)BB * CC * TT * 4);
  float*          vf   = (float*)take((size_t)BB * CC * TT * 4);
  float*          qf   = (float*)take((size_t)BB * CC * SS * 4);
  unsigned short* ktb  = (unsigned short*)take((size_t)BB * TT * CC * 2);
  unsigned short* qtb  = (unsigned short*)take((size_t)BB * SS * CC * 2);
  int*            nn   = (int*)take((size_t)BB * TT * KK * 4);

  auto cast = [&](const float* i, unsigned short* o, int n) {
    cast_f32_bf16<<<(n + 255) / 256, 256, 0, stream>>>(i, o, n);
  };
  cast(x,  xbf,  BB * CC * TT);
  cast(y,  ybf,  BB * CC * SS);
  cast(Wk, wkbf, TT * TT);
  cast(Wv, wvbf, TT * TT);
  cast(Wq, wqbf, SS * SS);

  // k = x*Wk^T, v = x*Wv^T : [2048,4096]x[4096,4096]^T  (TDM + WMMA)
  gemm_nt<<<dim3(TT / 64, (BB * CC) / 64), 128, 0, stream>>>(xbf, wkbf, kf, BB * CC, TT, TT);
  gemm_nt<<<dim3(TT / 64, (BB * CC) / 64), 128, 0, stream>>>(xbf, wvbf, vf, BB * CC, TT, TT);
  // q = y*Wq^T : [2048,512]x[512,512]^T
  gemm_nt<<<dim3(SS / 64, (BB * CC) / 64), 128, 0, stream>>>(ybf, wqbf, qf, BB * CC, SS, SS);

  // channel-norm + transpose for the similarity GEMM
  norm_tr<<<(BB * TT + 255) / 256, 256, 0, stream>>>(kf, ktb, TT);
  norm_tr<<<(BB * SS + 255) / 256, 256, 0, stream>>>(qf, qtb, SS);

  // fused cosine-sim (WMMA) + relu + top-8 + index remap
  sim_topk<<<dim3(TT / 16, BB), 32, 0, stream>>>(ktb, qtb, sidx, nn);

  // gather neighbors of v and apply the stride-9 conv + bias
  gather_conv<<<dim3(TT, BB), OCC, 0, stream>>>(vf, nn, cw, cb, out);
}
